// DualPathXF_69243462746789
// MI455X (gfx1250) — matrix operand order repaired and checked by reading, not compiled, not run on hardware
//
#include <hip/hip_runtime.h>

// MI455X / gfx1250 implementation of DualPathXF forward.
// All heavy GEMMs run through a wave32 WMMA kernel (v_wmma_f32_16x16x32_f16),
// template-specialized per epilogue so the inner loop carries no runtime flag
// branches. BN/ReLU/residual/f16-store fused in the epilogue. 3x3 conv = 9
// accumulating shifted GEMMs. Pixel<->memory attention = batched WMMA GEMMs
// (128 batches = B*NH) with rowwise softmax in between.
// NOTE: all GEMM N dimensions are multiples of 16, so the B-tile loader has
// no per-element tail path (16-column group is either fully valid or zero).

typedef _Float16 h16;
typedef _Float16 v16h __attribute__((ext_vector_type(16)));
typedef _Float16 v8h  __attribute__((ext_vector_type(8)));
typedef float    v8f  __attribute__((ext_vector_type(8)));

#define F_RELU   1
#define F_ACCUM  2
#define F_SHIFT  4
#define F_OUTF16 8
#define F_BN     16
#define F_RES    32

union ABfrag { v16h v; v8h h[2]; };

// ---------------------------------------------------------------------------
// Generic tiled WMMA GEMM: C = A(f16)[MxK] * B(f16)[KxN] (+ epilogue)
// 128 threads = 4 waves; 64x64 tile; BK=32; each wave owns a 32x32 subtile
// (2x2 v_wmma_f32_16x16x32_f16 accumulators). Software-pipelined k-loop:
// global->regs for tile k+1 overlaps WMMA on tile k.
// Batched via blockIdx.z with two-level (b, i) offsets.
// ---------------------------------------------------------------------------
template <int FL>
__global__ __launch_bounds__(128) void k_gemm(
    const h16* __restrict__ A, const h16* __restrict__ Bw,
    float* __restrict__ Cf, h16* __restrict__ Ch,
    const float* __restrict__ scale, const float* __restrict__ shiftv,
    const float* __restrict__ res,
    int M, int N, int Kd, int lda, int ldb, int ldc, int ldres,
    long long sAb, long long sAi, long long sBb, long long sBi,
    long long sCb, long long sCi, int iDiv, int dy, int dx)
{
  __shared__ __align__(16) h16 As[64][32];
  __shared__ __align__(16) h16 Bs[64][32];   // stored transposed: [n][k]
  int z  = blockIdx.z;
  int bz = z / iDiv, iz = z % iDiv;
  A  += (long long)bz * sAb + (long long)iz * sAi;
  Bw += (long long)bz * sBb + (long long)iz * sBi;
  long long cOff = (long long)bz * sCb + (long long)iz * sCi;

  int rowBase = blockIdx.y * 64, colBase = blockIdx.x * 64;
  int tid = threadIdx.x, lane = tid & 31;
  int wv = tid >> 5;
  int mW = (wv >> 1) * 32, nW = (wv & 1) * 32;
  int lo = lane & 15, hi = lane >> 4;

  // ---- staged-load helpers (global -> regs) ----
  int arow = tid >> 1, ach = tid & 1;          // A: 64 rows x 2 16-half chunks
  int agrow = rowBase + arow;
  bool avalid = agrow < M;
  long long asrow = agrow;
  if (FL & F_SHIFT) {
    int s = agrow % 1600;
    int y = s / 40, x = s - y * 40;
    int yy = y + dy, xx = x + dx;
    avalid = avalid && (yy >= 0) && (yy < 40) && (xx >= 0) && (xx < 40);
    asrow = agrow + dy * 40 + dx;
  }
  int bk = tid & 31, bng = tid >> 5;           // B: 32 k-rows x 4 16-col groups
  int bcol0 = colBase + bng * 16;
  bool bvalid = bcol0 < N;                     // N always a multiple of 16

  auto loadA = [&](int k0, v8h& a0, v8h& a1) {
    #pragma unroll
    for (int q = 0; q < 8; ++q) { a0[q] = (h16)0.f; a1[q] = (h16)0.f; }
    int kk = k0 + ach * 16;
    if (avalid && kk < Kd) {
      const h16* src = A + asrow * (long long)lda + kk;
      a0 = *(const v8h*)src;
      a1 = *(const v8h*)(src + 8);
      if (k0 + 64 < Kd) __builtin_prefetch(src + 64, 0, 0);
    }
  };
  auto loadB = [&](int k0, v8h& b0, v8h& b1) {
    #pragma unroll
    for (int q = 0; q < 8; ++q) { b0[q] = (h16)0.f; b1[q] = (h16)0.f; }
    int kk = k0 + bk;
    if (bvalid && kk < Kd) {
      const h16* src = Bw + (long long)kk * ldb + bcol0;
      b0 = *(const v8h*)src;
      b1 = *(const v8h*)(src + 8);
    }
  };

  v8f acc[2][2];
  #pragma unroll
  for (int mi = 0; mi < 2; ++mi)
    #pragma unroll
    for (int ni = 0; ni < 2; ++ni)
      #pragma unroll
      for (int r = 0; r < 8; ++r) acc[mi][ni][r] = 0.f;

  if (FL & F_ACCUM) {
    const float* Cin = Cf + cOff;
    #pragma unroll
    for (int mi = 0; mi < 2; ++mi)
      #pragma unroll
      for (int ni = 0; ni < 2; ++ni)
        #pragma unroll
        for (int r = 0; r < 8; ++r) {
          int gr = rowBase + mW + mi * 16 + r + hi * 8;
          int gc = colBase + nW + ni * 16 + lo;
          if (gr < M && gc < N) acc[mi][ni][r] = Cin[(long long)gr * ldc + gc];
        }
  }

  v8h a0, a1, b0, b1;
  loadA(0, a0, a1);
  loadB(0, b0, b1);

  for (int k0 = 0; k0 < Kd; k0 += 32) {
    // commit staged regs into LDS
    *(v8h*)&As[arow][ach * 16]     = a0;
    *(v8h*)&As[arow][ach * 16 + 8] = a1;
    #pragma unroll
    for (int q = 0; q < 8; ++q) {
      Bs[bng * 16 + q][bk]     = b0[q];
      Bs[bng * 16 + 8 + q][bk] = b1[q];
    }
    __syncthreads();
    // stage next tile while the matrix pipe works
    bool more = (k0 + 32) < Kd;
    v8h na0, na1, nb0, nb1;
    if (more) { loadA(k0 + 32, na0, na1); loadB(k0 + 32, nb0, nb1); }

    // ISA 7.12.2 fragment layouts:
    //  A: lanes 0-15 row lo, K chunks {hi?8-15:0-7, hi?24-31:16-23}
    //  B: lanes 0-15 col lo, K chunks {hi?8-15:0-7, hi?24-31:16-23}
    #pragma unroll
    for (int mi = 0; mi < 2; ++mi) {
      ABfrag a;
      const v8h* ar = (const v8h*)&As[mW + mi * 16 + lo][0];
      a.h[0] = ar[hi];
      a.h[1] = ar[2 + hi];
      #pragma unroll
      for (int ni = 0; ni < 2; ++ni) {
        ABfrag b;
        const v8h* br = (const v8h*)&Bs[nW + ni * 16 + lo][0];
        b.h[0] = br[hi];
        b.h[1] = br[2 + hi];
        acc[mi][ni] = __builtin_amdgcn_wmma_f32_16x16x32_f16(
            false, a.v, false, b.v, (short)0, acc[mi][ni], false, false);
      }
    }
    __syncthreads();
    if (more) { a0 = na0; a1 = na1; b0 = nb0; b1 = nb1; }
  }

  float* Co = Cf ? Cf + cOff : (float*)0;
  h16*   Ho = Ch ? Ch + cOff : (h16*)0;
  #pragma unroll
  for (int mi = 0; mi < 2; ++mi)
    #pragma unroll
    for (int ni = 0; ni < 2; ++ni)
      #pragma unroll
      for (int r = 0; r < 8; ++r) {
        int gr = rowBase + mW + mi * 16 + r + hi * 8;
        int gc = colBase + nW + ni * 16 + lo;
        if (gr < M && gc < N) {
          float v = acc[mi][ni][r];
          if (FL & F_BN)   v = v * scale[gc] + shiftv[gc];
          if (FL & F_RES)  v += res[(long long)gr * ldres + gc];
          if (FL & F_RELU) v = fmaxf(v, 0.f);
          if (FL & F_OUTF16) Ho[(long long)gr * ldc + gc] = (h16)v;
          else               Co[(long long)gr * ldc + gc] = v;
        }
      }
}

template <int FL>
static void launch_gemm(dim3 g, hipStream_t stream,
    const h16* A, const h16* Bw, float* Cf, h16* Ch,
    const float* sc, const float* sh, const float* res,
    int M, int N, int Kd, int lda, int ldb, int ldc, int ldres,
    long long sAb, long long sAi, long long sBb, long long sBi,
    long long sCb, long long sCi, int iDiv, int dy, int dx) {
  k_gemm<FL><<<g, dim3(128), 0, stream>>>(A, Bw, Cf, Ch, sc, sh, res,
      M, N, Kd, lda, ldb, ldc, ldres, sAb, sAi, sBb, sBi, sCb, sCi,
      iDiv, dy, dx);
}

// ---------------------------------------------------------------------------
// BN fold: scale = g*rsqrt(v+eps), shift = b - m*scale
// ---------------------------------------------------------------------------
__global__ void k_fold_bn(float* sc, float* sh, const float* g, const float* b,
                          const float* m, const float* v, int n) {
  int i = blockIdx.x * blockDim.x + threadIdx.x;
  if (i < n) {
    float s = g[i] * rsqrtf(v[i] + 1e-5f);
    sc[i] = s;
    sh[i] = b[i] - m[i] * s;
  }
}

// strided f32 -> row-major f16 pack (covers weight transposes)
__global__ void k_cvt(h16* dst, const float* src, long long rows, long long cols,
                      long long srs, long long scs) {
  long long idx = (long long)blockIdx.x * blockDim.x + threadIdx.x;
  if (idx >= rows * cols) return;
  long long r = idx / cols, c = idx - r * cols;
  dst[idx] = (h16)src[r * srs + c * scs];
}

// pixel NCHW f32 -> NHWC f16  (R = 16*1600 rows, 1024 cols)
__global__ void k_pix(const float* pix, h16* dst) {
  long long idx = (long long)blockIdx.x * blockDim.x + threadIdx.x;
  if (idx >= 26214400LL) return;
  int r = (int)(idx >> 10), c = (int)(idx & 1023);
  int b = r / 1600, s = r - b * 1600;
  dst[idx] = (h16)pix[(long long)b * 1638400 + (long long)c * 1600 + s];
}

// P = relu(C_bn + pixel_nchw)  -> P f32 NHWC + f16 copy
__global__ void k_pres(const float* C, const float* pix, float* Pf, h16* Ph) {
  long long idx = (long long)blockIdx.x * blockDim.x + threadIdx.x;
  if (idx >= 26214400LL) return;
  int r = (int)(idx >> 10), c = (int)(idx & 1023);
  int b = r / 1600, s = r - b * 1600;
  float v = fmaxf(C[idx] + pix[(long long)b * 1638400 + (long long)c * 1600 + s], 0.f);
  Pf[idx] = v;
  Ph[idx] = (h16)v;
}

// P_out = relu(C_bn + P) written NCHW to d_out
__global__ void k_finalp(const float* C, const float* Pf, float* outp) {
  long long idx = (long long)blockIdx.x * blockDim.x + threadIdx.x;
  if (idx >= 26214400LL) return;
  int r = (int)(idx >> 10), c = (int)(idx & 1023);
  int b = r / 1600, s = r - b * 1600;
  float v = fmaxf(C[idx] + Pf[idx], 0.f);
  outp[(long long)b * 1638400 + (long long)c * 1600 + s] = v;
}

// axial attention for one (batch, fixed-axis pos, head); L=40, DQ=16, DV=32.
// Reads qkv f32 [R,512] (BN already applied), writes interleaved y512:
// channel head*64 + 2c (= sv[c]) and head*64 + 2c+1 (= sve[c]).
__global__ __launch_bounds__(64) void k_axial(
    const float* __restrict__ qkv, float* __restrict__ y512,
    const float* __restrict__ pos, const float* __restrict__ aS,
    const float* __restrict__ aT, int axis)
{
  int blk = blockIdx.x;
  int head = blk & 7;
  int wp   = (blk >> 3) % 40;
  int b    = blk / 320;
  __shared__ float q[16][40], k[16][40], v[32][40], s[40][40];
  int t = threadIdx.x;

  for (int idx = t; idx < 64 * 40; idx += 64) {
    int c = idx / 40, l = idx - (idx / 40) * 40;
    int r = (axis == 0) ? (b * 1600 + l * 40 + wp) : (b * 1600 + wp * 40 + l);
    float val = qkv[(long long)r * 512 + head * 64 + c];
    if (c < 16)      q[c][l] = val;
    else if (c < 32) k[c - 16][l] = val;
    else             v[c - 32][l] = val;
  }
  __syncthreads();
  float sA = aS[head],      tA = aT[head];
  float sB = aS[8 + head],  tB = aT[8 + head];
  float sC = aS[16 + head], tC = aT[16 + head];
  for (int ij = t; ij < 1600; ij += 64) {
    int i = ij / 40, j = ij - (ij / 40) * 40;
    float qk = 0.f, qr = 0.f, kr = 0.f;
    int rij = i - j + 39, rji = j - i + 39;
    for (int c = 0; c < 16; ++c) {
      qk += q[c][i] * k[c][j];
      qr += q[c][i] * pos[c * 79 + rij];
      kr += k[c][j] * pos[(16 + c) * 79 + rji];
    }
    s[i][j] = (sA * qk + tA) + (sB * qr + tB) + (sC * kr + tC);
  }
  __syncthreads();
  for (int i = t; i < 40; i += 64) {
    float mx = -1e30f;
    for (int j = 0; j < 40; ++j) mx = fmaxf(mx, s[i][j]);
    float sum = 0.f;
    for (int j = 0; j < 40; ++j) { float e = __expf(s[i][j] - mx); s[i][j] = e; sum += e; }
    float inv = 1.f / sum;
    for (int j = 0; j < 40; ++j) s[i][j] *= inv;
  }
  __syncthreads();
  for (int ci = t; ci < 64 * 40; ci += 64) {
    int c = ci / 40, i = ci - (ci / 40) * 40;
    int cc = c >> 1;
    float a2 = 0.f;
    if ((c & 1) == 0) { for (int j = 0; j < 40; ++j) a2 += s[i][j] * v[cc][j]; }
    else              { for (int j = 0; j < 40; ++j) a2 += s[i][j] * pos[(32 + cc) * 79 + (i - j + 39)]; }
    int r = (axis == 0) ? (b * 1600 + i * 40 + wp) : (b * 1600 + wp * 40 + i);
    y512[(long long)r * 512 + head * 64 + c] = a2;
  }
}

// bn_out over 512 channels + pair-sum to 256 (+ optional relu), f16 out
__global__ void k_pairsum(const float* y512, const float* sc, const float* sh,
                          h16* out, int relu, long long total) {
  long long idx = (long long)blockIdx.x * blockDim.x + threadIdx.x;
  if (idx >= total) return;
  int r = (int)(idx >> 8), c = (int)(idx & 255);
  const float* yr = y512 + (long long)r * 512;
  float a = yr[2 * c] * sc[2 * c] + sh[2 * c];
  float b = yr[2 * c + 1] * sc[2 * c + 1] + sh[2 * c + 1];
  float v = a + b;
  if (relu) v = fmaxf(v, 0.f);
  out[idx] = (h16)v;
}

// repack memory qkv heads:  Mq f32 [2048,512] -> qm[b,i,128,16] f16,
// km[b,i,16,128] f16 (transposed), vm[b,i,128,32] f16
__global__ __launch_bounds__(64) void k_memheads(const float* Mq, h16* qm,
                                                 h16* km, h16* vm) {
  int blk = blockIdx.x;
  int n = blk & 127, i = (blk >> 7) & 7, b = blk >> 10;
  int j = threadIdx.x;
  float v = Mq[((long long)(n * 16 + b)) * 512 + i * 64 + j];
  int bi = b * 8 + i;
  if (j < 16)       qm[((long long)bi * 128 + n) * 16 + j] = (h16)v;
  else if (j < 32)  km[((long long)bi * 16 + (j - 16)) * 128 + n] = (h16)v;
  else              vm[((long long)bi * 128 + n) * 32 + (j - 32)] = (h16)v;
}

// build kpm[b,i,16,1728] and vpm[b,i,1728,32] from pixel qkv + memory qkv
__global__ __launch_bounds__(64) void k_kpmvpm(const float* PQ, const float* Mq,
                                               h16* kpm, h16* vpm) {
  int blk = blockIdx.x;
  int l  = blk % 1728;
  int bi = blk / 1728;
  int b = bi >> 3, i = bi & 7;
  int t = threadIdx.x;
  const float* row;
  if (l < 1600) row = PQ + ((long long)b * 1600 + l) * 512 + i * 64;
  else          row = Mq + ((long long)(l - 1600) * 16 + b) * 512 + i * 64;
  if (t < 16)      kpm[((long long)bi * 16 + t) * 1728 + l] = (h16)row[16 + t];
  else if (t < 48) vpm[((long long)bi * 1728 + l) * 32 + (t - 16)] = (h16)row[32 + (t - 16)];
}

// rowwise softmax, f32 in -> normalized f16 out (feeds next WMMA GEMM)
__global__ __launch_bounds__(256) void k_softmax(const float* in, h16* out, int n) {
  __shared__ float red[256];
  long long row = blockIdx.x;
  in  += row * (long long)n;
  out += row * (long long)n;
  int t = threadIdx.x;
  float mx = -1e30f;
  for (int j = t; j < n; j += 256) mx = fmaxf(mx, in[j]);
  red[t] = mx; __syncthreads();
  for (int s = 128; s > 0; s >>= 1) { if (t < s) red[t] = fmaxf(red[t], red[t + s]); __syncthreads(); }
  mx = red[0]; __syncthreads();
  float sum = 0.f;
  for (int j = t; j < n; j += 256) sum += __expf(in[j] - mx);
  red[t] = sum; __syncthreads();
  for (int s = 128; s > 0; s >>= 1) { if (t < s) red[t] += red[t + s]; __syncthreads(); }
  float inv = 1.f / red[0];
  for (int j = t; j < n; j += 256) out[j] = (h16)(__expf(in[j] - mx) * inv);
}

// ---------------------------------------------------------------------------
extern "C" void kernel_launch(void* const* d_in, const int* in_sizes, int n_in,
                              void* d_out, int out_size, void* d_ws, size_t ws_size,
                              hipStream_t stream) {
  (void)in_sizes; (void)n_in; (void)out_size; (void)ws_size;
  const long long R = 25600;           // B*H*W = 16*40*40
  #define IN(i) ((const float*)d_in[i])

  char* ws = (char*)d_ws;
  size_t off = 0;
  auto alloc = [&](size_t bytes) -> void* {
    void* p = ws + off;
    off += (bytes + 255) & ~(size_t)255;
    return p;
  };

  // f16 weight packs
  h16* wc1   = (h16*)alloc(1024 * 256 * 2);
  h16* wqh   = (h16*)alloc(256 * 512 * 2);
  h16* wqw   = (h16*)alloc(256 * 512 * 2);
  h16* wc2   = (h16*)alloc(256 * 1024 * 2);
  h16* wpm1  = (h16*)alloc(1024 * 256 * 2);
  h16* wqkv9 = (h16*)alloc(9 * 256 * 512 * 2);
  h16* wpm2  = (h16*)alloc(256 * 1024 * 2);
  h16* wfc1  = (h16*)alloc(256 * 256 * 2);
  h16* wmq   = (h16*)alloc(256 * 512 * 2);
  h16* wfc2  = (h16*)alloc(256 * 256 * 2);
  h16* wff1  = (h16*)alloc(256 * 1024 * 2);
  h16* wff2  = (h16*)alloc(1024 * 256 * 2);
  // folded BN params (scale, shift pairs)
  auto bnpair = [&](int n, float** sc, float** sh) {
    *sc = (float*)alloc(n * 4); *sh = (float*)alloc(n * 4);
  };
  float *sc_c1,*sh_c1, *sc_hq,*sh_hq, *sc_ha,*sh_ha, *sc_ho,*sh_ho;
  float *sc_wq,*sh_wq, *sc_wa,*sh_wa, *sc_wo,*sh_wo, *sc_c2,*sh_c2;
  float *sc_p1,*sh_p1, *sc_pq,*sh_pq, *sc_p2,*sh_p2;
  float *sc_f1,*sh_f1, *sc_mq,*sh_mq, *sc_f2,*sh_f2, *sc_g1,*sh_g1, *sc_g2,*sh_g2;
  bnpair(256,&sc_c1,&sh_c1);  bnpair(512,&sc_hq,&sh_hq);  bnpair(24,&sc_ha,&sh_ha);
  bnpair(512,&sc_ho,&sh_ho);  bnpair(512,&sc_wq,&sh_wq);  bnpair(24,&sc_wa,&sh_wa);
  bnpair(512,&sc_wo,&sh_wo);  bnpair(1024,&sc_c2,&sh_c2); bnpair(256,&sc_p1,&sh_p1);
  bnpair(512,&sc_pq,&sh_pq);  bnpair(1024,&sc_p2,&sh_p2); bnpair(256,&sc_f1,&sh_f1);
  bnpair(512,&sc_mq,&sh_mq);  bnpair(256,&sc_f2,&sh_f2);  bnpair(1024,&sc_g1,&sh_g1);
  bnpair(256,&sc_g2,&sh_g2);

  // activations
  h16*   pix16 = (h16*)alloc(R * 1024 * 2);
  h16*   X1    = (h16*)alloc(R * 256 * 2);
  float* PQKV  = (float*)alloc(R * 512 * 4);
  float* Y512  = (float*)alloc(R * 512 * 4);
  h16*   YH    = (h16*)alloc(R * 256 * 2);
  h16*   YW    = (h16*)alloc(R * 256 * 2);
  float* BIGC  = (float*)alloc(R * 1024 * 4);     // conv2 C, later p2m_conv2 C
  float* PF    = (float*)alloc(R * 1024 * 4);
  h16*   P16   = (h16*)alloc(R * 1024 * 2);
  h16*   PC1   = (h16*)alloc(R * 256 * 2);
  float* PQKV2 = (float*)alloc(R * 512 * 4);
  h16*   PQ16  = (h16*)alloc(R * 512 * 2);
  h16*   MEM16 = (h16*)alloc(2048 * 256 * 2);
  h16*   MF1   = (h16*)alloc(2048 * 256 * 2);
  float* MQ    = (float*)alloc(2048 * 512 * 4);
  h16*   QM16  = (h16*)alloc((size_t)128 * 128 * 16 * 2);
  h16*   KM16  = (h16*)alloc((size_t)128 * 16 * 128 * 2);
  h16*   VM16  = (h16*)alloc((size_t)128 * 128 * 32 * 2);
  float* SBUF  = (float*)alloc((size_t)128 * 128 * 1728 * 4);   // reused: scores then m2
  h16*   SSOFT = (h16*)alloc((size_t)128 * 128 * 1728 * 2);
  float* YPA   = (float*)alloc(R * 256 * 4);
  h16*   YPA16 = (h16*)alloc(R * 256 * 2);
  h16*   KPM16 = (h16*)alloc((size_t)128 * 16 * 1728 * 2);
  h16*   VPM16 = (h16*)alloc((size_t)128 * 1728 * 32 * 2);
  float* YMB   = (float*)alloc(2048 * 256 * 4);
  h16*   YMB16 = (h16*)alloc(2048 * 256 * 2);
  float* M1    = (float*)alloc(2048 * 256 * 4);
  h16*   M1_16 = (h16*)alloc(2048 * 256 * 2);
  h16*   FF1   = (h16*)alloc(2048 * 1024 * 2);

  float* outP = (float*)d_out;
  float* outM = outP + 26214400LL;

  auto fold = [&](float* sc, float* sh, int gi, int n) {
    k_fold_bn<<<dim3((n + 63) / 64), dim3(64), 0, stream>>>(
        sc, sh, IN(gi), IN(gi + 1), IN(gi + 2), IN(gi + 3), n);
  };
  auto pack = [&](h16* dst, const float* src, long long rows, long long cols,
                  long long srs, long long scs) {
    long long tot = rows * cols;
    k_cvt<<<dim3((unsigned)((tot + 255) / 256)), dim3(256), 0, stream>>>(
        dst, src, rows, cols, srs, scs);
  };
  auto gemm = [&](const h16* A, const h16* Bw, float* Cf, h16* Ch,
                  const float* sc, const float* sh, const float* res,
                  int M, int N, int Kd, int lda, int ldb, int ldc, int ldres,
                  long long sAb, long long sAi, long long sBb, long long sBi,
                  long long sCb, long long sCi, int iDiv, int nb,
                  int flags, int dy, int dx) {
    dim3 g((N + 63) / 64, (M + 63) / 64, nb);
    int id = iDiv < 1 ? 1 : iDiv;
    switch (flags) {
      case 0:
        launch_gemm<0>(g, stream, A, Bw, Cf, Ch, sc, sh, res, M, N, Kd,
            lda, ldb, ldc, ldres, sAb, sAi, sBb, sBi, sCb, sCi, id, dy, dx); break;
      case F_BN:
        launch_gemm<F_BN>(g, stream, A, Bw, Cf, Ch, sc, sh, res, M, N, Kd,
            lda, ldb, ldc, ldres, sAb, sAi, sBb, sBi, sCb, sCi, id, dy, dx); break;
      case F_OUTF16 | F_BN | F_RELU:
        launch_gemm<F_OUTF16 | F_BN | F_RELU>(g, stream, A, Bw, Cf, Ch, sc, sh, res,
            M, N, Kd, lda, ldb, ldc, ldres, sAb, sAi, sBb, sBi, sCb, sCi, id, dy, dx); break;
      case F_SHIFT:
        launch_gemm<F_SHIFT>(g, stream, A, Bw, Cf, Ch, sc, sh, res, M, N, Kd,
            lda, ldb, ldc, ldres, sAb, sAi, sBb, sBi, sCb, sCi, id, dy, dx); break;
      case F_SHIFT | F_ACCUM:
        launch_gemm<F_SHIFT | F_ACCUM>(g, stream, A, Bw, Cf, Ch, sc, sh, res,
            M, N, Kd, lda, ldb, ldc, ldres, sAb, sAi, sBb, sBi, sCb, sCi, id, dy, dx); break;
      case F_SHIFT | F_ACCUM | F_BN:
        launch_gemm<F_SHIFT | F_ACCUM | F_BN>(g, stream, A, Bw, Cf, Ch, sc, sh, res,
            M, N, Kd, lda, ldb, ldc, ldres, sAb, sAi, sBb, sBi, sCb, sCi, id, dy, dx); break;
      case F_BN | F_RES | F_RELU:
        launch_gemm<F_BN | F_RES | F_RELU>(g, stream, A, Bw, Cf, Ch, sc, sh, res,
            M, N, Kd, lda, ldb, ldc, ldres, sAb, sAi, sBb, sBi, sCb, sCi, id, dy, dx); break;
      default: break;
    }
  };

  // ---- fold all BNs ----
  fold(sc_c1, sh_c1, 3, 256);    fold(sc_hq, sh_hq, 8, 512);
  fold(sc_ha, sh_ha, 12, 24);    fold(sc_ho, sh_ho, 16, 512);
  fold(sc_wq, sh_wq, 22, 512);   fold(sc_wa, sh_wa, 26, 24);
  fold(sc_wo, sh_wo, 30, 512);   fold(sc_c2, sh_c2, 36, 1024);
  fold(sc_p1, sh_p1, 41, 256);   fold(sc_pq, sh_pq, 46, 512);
  fold(sc_p2, sh_p2, 51, 1024);  fold(sc_f1, sh_f1, 56, 256);
  fold(sc_mq, sh_mq, 61, 512);   fold(sc_f2, sh_f2, 66, 256);
  fold(sc_g1, sh_g1, 71, 1024);  fold(sc_g2, sh_g2, 76, 256);

  // ---- weight / input packs (B matrices stored [K][N]) ----
  k_pix<<<dim3(102400), dim3(256), 0, stream>>>(IN(0), pix16);
  pack(wc1,  IN(2), 1024, 256, 1, 1024);
  pack(wqh,  IN(7), 256, 512, 1, 256);
  pack(wqw,  IN(21), 256, 512, 1, 256);
  pack(wc2,  IN(35), 256, 1024, 1, 256);
  pack(wpm1, IN(40), 1024, 256, 1, 1024);
  for (int t = 0; t < 9; ++t)
    pack(wqkv9 + (size_t)t * 256 * 512, IN(45) + t, 256, 512, 9, 2304);
  pack(wpm2, IN(50), 256, 1024, 1, 256);
  pack(wfc1, IN(55), 256, 256, 1, 256);
  pack(wmq,  IN(60), 256, 512, 1, 256);
  pack(wfc2, IN(65), 256, 256, 1, 256);
  pack(wff1, IN(70), 256, 1024, 1, 256);
  pack(wff2, IN(75), 1024, 256, 1, 1024);
  pack(MEM16, IN(1), 2048, 256, 256, 1);

  // ---- p2p pixel path ----
  gemm(pix16, wc1, nullptr, X1, sc_c1, sh_c1, nullptr,
       (int)R, 256, 1024, 1024, 256, 256, 0, 0,0,0,0,0,0, 1, 1,
       F_OUTF16 | F_BN | F_RELU, 0, 0);
  gemm(X1, wqh, PQKV, nullptr, sc_hq, sh_hq, nullptr,
       (int)R, 512, 256, 256, 512, 512, 0, 0,0,0,0,0,0, 1, 1, F_BN, 0, 0);
  k_axial<<<dim3(5120), dim3(64), 0, stream>>>(PQKV, Y512, IN(20), sc_ha, sh_ha, 0);
  k_pairsum<<<dim3(25600), dim3(256), 0, stream>>>(Y512, sc_ho, sh_ho, YH, 0, R * 256);
  gemm(YH, wqw, PQKV, nullptr, sc_wq, sh_wq, nullptr,
       (int)R, 512, 256, 256, 512, 512, 0, 0,0,0,0,0,0, 1, 1, F_BN, 0, 0);
  k_axial<<<dim3(5120), dim3(64), 0, stream>>>(PQKV, Y512, IN(34), sc_wa, sh_wa, 1);
  k_pairsum<<<dim3(25600), dim3(256), 0, stream>>>(Y512, sc_wo, sh_wo, YW, 1, R * 256);
  gemm(YW, wc2, BIGC, nullptr, sc_c2, sh_c2, nullptr,
       (int)R, 1024, 256, 256, 1024, 1024, 0, 0,0,0,0,0,0, 1, 1, F_BN, 0, 0);
  k_pres<<<dim3(102400), dim3(256), 0, stream>>>(BIGC, IN(0), PF, P16);

  // ---- p2m pixel qkv: 1x1 conv then 3x3 conv as 9 shifted GEMMs ----
  gemm(P16, wpm1, nullptr, PC1, sc_p1, sh_p1, nullptr,
       (int)R, 256, 1024, 1024, 256, 256, 0, 0,0,0,0,0,0, 1, 1,
       F_OUTF16 | F_BN | F_RELU, 0, 0);
  for (int t = 0; t < 9; ++t) {
    int ky = t / 3, kx = t % 3;
    int fl = F_SHIFT | (t ? F_ACCUM : 0) | (t == 8 ? F_BN : 0);
    gemm(PC1, wqkv9 + (size_t)t * 256 * 512, PQKV2, nullptr, sc_pq, sh_pq, nullptr,
         (int)R, 512, 256, 256, 512, 512, 0, 0,0,0,0,0,0, 1, 1, fl, ky - 1, kx - 1);
  }
  pack(PQ16, PQKV2, R, 512, 512, 1);

  // ---- memory qkv ----
  gemm(MEM16, wfc1, nullptr, MF1, sc_f1, sh_f1, nullptr,
       2048, 256, 256, 256, 256, 256, 0, 0,0,0,0,0,0, 1, 1,
       F_OUTF16 | F_BN | F_RELU, 0, 0);
  gemm(MF1, wmq, MQ, nullptr, sc_mq, sh_mq, nullptr,
       2048, 512, 256, 256, 512, 512, 0, 0,0,0,0,0,0, 1, 1, F_BN, 0, 0);
  k_memheads<<<dim3(16384), dim3(64), 0, stream>>>(MQ, QM16, KM16, VM16);

  // ---- pixel <- memory attention (batched per (b,head): 128 batches) ----
  gemm(PQ16, KM16, SBUF, nullptr, nullptr, nullptr, nullptr,
       1600, 128, 16, 512, 128, 128, 0,
       1600LL * 512, 64, 8LL * 2048, 2048, 8LL * 204800, 204800, 8, 128, 0, 0, 0);
  k_softmax<<<dim3(204800), dim3(256), 0, stream>>>(SBUF, SSOFT, 128);
  gemm(SSOFT, VM16, YPA, nullptr, nullptr, nullptr, nullptr,
       1600, 32, 128, 128, 32, 256, 0,
       8LL * 204800, 204800, 8LL * 4096, 4096, 1600LL * 256, 32, 8, 128, 0, 0, 0);
  pack(YPA16, YPA, R, 256, 256, 1);
  gemm(YPA16, wpm2, BIGC, nullptr, sc_p2, sh_p2, nullptr,
       (int)R, 1024, 256, 256, 1024, 1024, 0, 0,0,0,0,0,0, 1, 1, F_BN, 0, 0);
  k_finalp<<<dim3(102400), dim3(256), 0, stream>>>(BIGC, PF, outP);

  // ---- memory <- (pixel + memory) attention ----
  k_kpmvpm<<<dim3(221184), dim3(64), 0, stream>>>(PQKV2, MQ, KPM16, VPM16);
  gemm(QM16, KPM16, SBUF, nullptr, nullptr, nullptr, nullptr,
       128, 1728, 16, 16, 1728, 1728, 0,
       8LL * 2048, 2048, 8LL * 27648, 27648, 8LL * 221184, 221184, 8, 128, 0, 0, 0);
  k_softmax<<<dim3(16384), dim3(256), 0, stream>>>(SBUF, SSOFT, 1728);
  gemm(SSOFT, VPM16, YMB, nullptr, nullptr, nullptr, nullptr,
       128, 32, 1728, 1728, 32, 4096, 0,
       8LL * 221184, 221184, 8LL * 55296, 55296, 256, 32, 8, 128, 0, 0, 0);
  pack(YMB16, YMB, 2048, 256, 256, 1);
  gemm(YMB16, wfc2, M1, nullptr, sc_f2, sh_f2, IN(1),
       2048, 256, 256, 256, 256, 256, 256, 0,0,0,0,0,0, 1, 1,
       F_BN | F_RES | F_RELU, 0, 0);
  pack(M1_16, M1, 2048, 256, 256, 1);
  gemm(M1_16, wff1, nullptr, FF1, sc_g1, sh_g1, nullptr,
       2048, 1024, 256, 256, 1024, 1024, 0, 0,0,0,0,0,0, 1, 1,
       F_OUTF16 | F_BN | F_RELU, 0, 0);
  gemm(FF1, wff2, outM, nullptr, sc_g2, sh_g2, M1,
       2048, 256, 1024, 1024, 256, 256, 256, 0,0,0,0,0,0, 1, 1,
       F_BN | F_RES | F_RELU, 0, 0);
  #undef IN
}
